// PreT_Attention_15908558864722
// MI455X (gfx1250) — matrix operand order repaired and checked by reading, not compiled, not run on hardware
//
#include <hip/hip_runtime.h>

// ---------------------------------------------------------------------------
// Types for CDNA5 WMMA (wave32, v_wmma_f32_16x16x32_bf16)
// ---------------------------------------------------------------------------
typedef __bf16 bf16_t;
typedef bf16_t v8bf  __attribute__((ext_vector_type(8)));
typedef bf16_t v16bf __attribute__((ext_vector_type(16)));
typedef float  v8f   __attribute__((ext_vector_type(8)));

// A-fragment (16x32 bf16, MxK). Lanes 0-15 -> M=0..15, K=0..15;
// lanes 16-31 -> same M rows, K=16..31 (contiguous, matching load_fragB).
// Per lane this is one contiguous 16-element (32B) run at kb+g*16.
__device__ __forceinline__ v16bf load_fragA(const bf16_t* base, long row,
                                            int lda, int kb, int lane) {
  int g = lane >> 4;
  return *(const v16bf*)(base + row * (long)lda + kb + g * 16);
}

// B-fragment (32x16 bf16, KxN), column j stored row-major with K contiguous
// (NT layout). Lane l: N = l&15, K = kb + (l>>4)*16 .. +15 -> one 32B run.
__device__ __forceinline__ v16bf load_fragB(const bf16_t* base, long col,
                                            int ldb, int kb, int lane) {
  int g = lane >> 4;
  return *(const v16bf*)(base + col * (long)ldb + kb + g * 16);
}

__device__ __forceinline__ v8f wmma_bf16(v16bf a, v16bf b, v8f c) {
  return __builtin_amdgcn_wmma_f32_16x16x32_bf16(false, a, false, b,
                                                 (short)0, c, false, false);
}

// ---------------------------------------------------------------------------
// f32 -> bf16 conversion
// ---------------------------------------------------------------------------
__global__ __launch_bounds__(256) void cvt_bf16(const float* __restrict__ in,
                                                bf16_t* __restrict__ out, int n) {
  int i = blockIdx.x * 256 + threadIdx.x;
  if (i < n) out[i] = (bf16_t)in[i];
}

// ---------------------------------------------------------------------------
// Generic NT GEMM: C[MxN] = A[MxK] * B[NxK]^T  (+bias).  One wave = 64x64 tile.
// Cf != nullptr -> f32 output (+bias);  else bf16 output.
// M % 64 == 0, N % 64 == 0, K % 32 == 0.
// ---------------------------------------------------------------------------
__global__ __launch_bounds__(256)
void gemm_nt_wmma(const bf16_t* __restrict__ A, const bf16_t* __restrict__ B,
                  int M, int N, int K, bf16_t* __restrict__ Cb,
                  float* __restrict__ Cf, const float* __restrict__ bias) {
  const int lane = threadIdx.x & 31;
  const int wave = threadIdx.x >> 5;
  const int tilesN = N >> 6;
  const int totalTiles = (M >> 6) * tilesN;
  const int tile = blockIdx.x * 8 + wave;
  if (tile >= totalTiles) return;               // wave-uniform: EXEC stays full
  const int tm = (tile / tilesN) << 6;
  const int tn = (tile % tilesN) << 6;

  v8f acc[4][4] = {};
  const int rA = lane & 15;
  for (int kb = 0; kb < K; kb += 32) {
    v16bf a[4], b[4];
#pragma unroll
    for (int i = 0; i < 4; ++i)
      a[i] = load_fragA(A, (long)(tm + i * 16 + rA), K, kb, lane);
#pragma unroll
    for (int j = 0; j < 4; ++j)
      b[j] = load_fragB(B, (long)(tn + j * 16 + rA), K, kb, lane);
#pragma unroll
    for (int i = 0; i < 4; ++i)
#pragma unroll
      for (int j = 0; j < 4; ++j) acc[i][j] = wmma_bf16(a[i], b[j], acc[i][j]);
  }

  // D layout: lane l -> N = l&15, rows M = (l>>4)*8 + v
  const int col = lane & 15;
  const int rb8 = (lane >> 4) * 8;
#pragma unroll
  for (int i = 0; i < 4; ++i)
#pragma unroll
    for (int j = 0; j < 4; ++j)
#pragma unroll
      for (int v = 0; v < 8; ++v) {
        long m = tm + i * 16 + rb8 + v;
        long n = tn + j * 16 + col;
        float val = acc[i][j][v];
        if (Cf) Cf[m * N + n] = val + (bias ? bias[n] : 0.0f);
        else    Cb[m * N + n] = (bf16_t)val;
      }
}

// ---------------------------------------------------------------------------
// Fused attention: one workgroup per (b, head).  S (208x208 f32), P (208x224
// bf16), vT (64x224 bf16) in one dynamic-LDS slab (288 KB; CDNA5 WG max 320KB).
// qkvb layout: row = b*197+n (stride 2304), cols = which*768 + h*64 + d.
// ---------------------------------------------------------------------------
#define SM_S_BYTES (208 * 208 * 4)
#define SM_P_BYTES (208 * 224 * 2)
#define SM_V_BYTES (64 * 224 * 2)
#define SM_TOTAL   (SM_S_BYTES + SM_P_BYTES + SM_V_BYTES)

__global__ __launch_bounds__(256)
void attn_fused(const bf16_t* __restrict__ qkvb, bf16_t* __restrict__ ob) {
  extern __shared__ char smem[];
  float*  S  = (float*)smem;                                  // 208 x 208
  bf16_t* P  = (bf16_t*)(smem + SM_S_BYTES);                  // 208 x 224
  bf16_t* vT = (bf16_t*)(smem + SM_S_BYTES + SM_P_BYTES);     // 64  x 224

  const int b = blockIdx.x, h = blockIdx.y;
  const int tid = threadIdx.x, lane = tid & 31, wave = tid >> 5;
  const long rowbase = (long)b * 197 * 2304;
  const bf16_t* qbase = qkvb + rowbase + h * 64;
  const bf16_t* kbase = qkvb + rowbase + 768 + h * 64;
  const bf16_t* vbase = qkvb + rowbase + 1536 + h * 64;

  // stage v transposed (d-major, K=j contiguous), zero-pad j>=197
  for (int idx = tid; idx < 64 * 224; idx += 256) {
    int d = idx / 224, j = idx % 224;
    vT[d * 224 + j] = (j < 197) ? vbase[(long)j * 2304 + d] : (bf16_t)0.0f;
  }

  // ---- phase 1: S = q @ k^T  (13x13 tiles of 16x16, K=64) ----
  const int rA = lane & 15;
  for (int t = wave; t < 13 * 13; t += 8) {
    int ti = t / 13, tj = t % 13;
    int ra = ti * 16 + rA; if (ra > 196) ra = 196;   // clamp pad rows
    int rb = tj * 16 + rA; if (rb > 196) rb = 196;
    v8f acc = {};
#pragma unroll
    for (int kb = 0; kb < 64; kb += 32) {
      v16bf a  = load_fragA(qbase, (long)ra, 2304, kb, lane);
      v16bf bb = load_fragB(kbase, (long)rb, 2304, kb, lane);
      acc = wmma_bf16(a, bb, acc);
    }
    int col = lane & 15, rb8 = (lane >> 4) * 8;
#pragma unroll
    for (int v = 0; v < 8; ++v)
      S[(ti * 16 + rb8 + v) * 208 + tj * 16 + col] = acc[v];
  }
  __syncthreads();

  // ---- phase 2: row softmax (scale = hd^-0.5 = 0.125), write bf16 P ----
  const float scale = 0.125f;
  for (int r = tid; r < 208; r += 256) {
    if (r < 197) {
      float m = -1e30f;
      for (int c = 0; c < 197; ++c) m = fmaxf(m, S[r * 208 + c]);
      float sum = 0.0f;
      for (int c = 0; c < 197; ++c) sum += __expf((S[r * 208 + c] - m) * scale);
      float inv = 1.0f / sum;
      for (int c = 0; c < 197; ++c)
        P[r * 224 + c] = (bf16_t)(__expf((S[r * 208 + c] - m) * scale) * inv);
      for (int c = 197; c < 224; ++c) P[r * 224 + c] = (bf16_t)0.0f;
    } else {
      for (int c = 0; c < 224; ++c) P[r * 224 + c] = (bf16_t)0.0f;
    }
  }
  __syncthreads();

  // ---- phase 3: O = P @ v  (13x4 tiles, K=224) ----
  for (int t = wave; t < 13 * 4; t += 8) {
    int ti = t >> 2, tj = t & 3;
    v8f acc = {};
#pragma unroll
    for (int kb = 0; kb < 224; kb += 32) {
      v16bf a  = load_fragA(P, (long)(ti * 16 + rA), 224, kb, lane);
      v16bf bb = load_fragB(vT, (long)(tj * 16 + rA), 224, kb, lane);
      acc = wmma_bf16(a, bb, acc);
    }
    int col = lane & 15, rb8 = (lane >> 4) * 8;
#pragma unroll
    for (int v = 0; v < 8; ++v) {
      int n = ti * 16 + rb8 + v;
      if (n < 197)
        ob[((long)b * 197 + n) * 768 + h * 64 + tj * 16 + col] = (bf16_t)acc[v];
    }
  }
}

// ---------------------------------------------------------------------------
// Token branch K1: per (s,b) selector cross-attention -> task_x rows (bf16)
// ---------------------------------------------------------------------------
__global__ __launch_bounds__(256)
void selector_attn(const float* __restrict__ x, const float* __restrict__ tokens,
                   bf16_t* __restrict__ txb) {
  const int s = blockIdx.x, b = blockIdx.y, tid = threadIdx.x;
  __shared__ float g[256];
  __shared__ float red[256];
  const float* sel = tokens + ((long)b * 9 + 1 + s) * 768;
  const float* xb = x + (long)b * 197 * 768;

  float gv = -1e30f;
  if (tid < 197) {
    float acc = 0.0f;
    const float* xr = xb + (long)tid * 768;
    for (int c = 0; c < 768; ++c) acc += sel[c] * xr[c];
    gv = acc * 0.03608439182435161f;  // 768^-0.5
    g[tid] = gv;
  }
  red[tid] = gv;
  __syncthreads();
  for (int off = 128; off > 0; off >>= 1) {
    if (tid < off) red[tid] = fmaxf(red[tid], red[tid + off]);
    __syncthreads();
  }
  float m = red[0];
  __syncthreads();
  float ev = (tid < 197) ? __expf(gv - m) : 0.0f;
  red[tid] = ev;
  __syncthreads();
  for (int off = 128; off > 0; off >>= 1) {
    if (tid < off) red[tid] += red[tid + off];
    __syncthreads();
  }
  float inv = 1.0f / red[0];
  __syncthreads();
  if (tid < 197) g[tid] = ev * inv;
  __syncthreads();

  for (int c = tid; c < 768; c += 256) {
    float acc = 0.0f;
    for (int n = 0; n < 197; ++n) acc += g[n] * xb[(long)n * 768 + c];
    txb[((long)b * 9 + 1 + s) * 768 + c] = (bf16_t)acc;
    if (s == 0)  // also write the cls row once
      txb[((long)b * 9) * 768 + c] = (bf16_t)tokens[((long)b * 9) * 768 + c];
  }
}

// ---------------------------------------------------------------------------
// Token branch K3: tiny prompt attention, cls row only -> out0[b, 768] (f32)
// prompts layout (2,10,8,12,64); T_TASK = 0.
// ---------------------------------------------------------------------------
__global__ __launch_bounds__(256)
void task_attn(const bf16_t* __restrict__ tq, const float* __restrict__ prompts,
               float* __restrict__ out0) {
  const int b = blockIdx.x, tid = threadIdx.x;
  __shared__ float Sg[12][17];
  __shared__ float Pp[12][17];
  if (tid < 204) {
    int h = tid / 17, j = tid % 17;
    const bf16_t* q0 = tq + ((long)b * 9) * 2304 + h * 64;
    float acc = 0.0f;
    if (j < 8) {
      const float* kp = prompts + ((long)j * 12 + h) * 64;  // [0,0,j,h,:]
      for (int d = 0; d < 64; ++d) acc += (float)q0[d] * kp[d];
    } else {
      const bf16_t* kr = tq + ((long)b * 9 + (j - 8)) * 2304 + 768 + h * 64;
      for (int d = 0; d < 64; ++d) acc += (float)q0[d] * (float)kr[d];
    }
    Sg[h][j] = acc * 0.125f;
  }
  __syncthreads();
  if (tid < 12) {
    float m = -1e30f;
    for (int j = 0; j < 17; ++j) m = fmaxf(m, Sg[tid][j]);
    float sum = 0.0f;
    for (int j = 0; j < 17; ++j) { float e = __expf(Sg[tid][j] - m); Pp[tid][j] = e; sum += e; }
    float inv = 1.0f / sum;
    for (int j = 0; j < 17; ++j) Pp[tid][j] *= inv;
  }
  __syncthreads();
  for (int idx = tid; idx < 768; idx += 256) {
    int h = idx >> 6, d = idx & 63;
    float acc = 0.0f;
    for (int j = 0; j < 17; ++j) {
      float vv;
      if (j < 8) vv = prompts[((long)(80 + j) * 12 + h) * 64 + d];  // [1,0,j,h,:]
      else vv = (float)tq[((long)b * 9 + (j - 8)) * 2304 + 1536 + h * 64 + d];
      acc += Pp[h][j] * vv;
    }
    out0[(long)b * 768 + idx] = acc;
  }
}

// Token branch K4: task_out cls row = out0 @ proj_w^T + proj_b (f32)
__global__ __launch_bounds__(256)
void task_proj(const float* __restrict__ out0, const float* __restrict__ proj_w,
               const float* __restrict__ proj_b, float* __restrict__ outTok) {
  const int b = blockIdx.x;
  const int col = blockIdx.y * 256 + threadIdx.x;
  if (col >= 768) return;
  float acc = proj_b[col];
  const float* o = out0 + (long)b * 768;
  const float* w = proj_w + (long)col * 768;
  for (int k = 0; k < 768; ++k) acc += o[k] * w[k];
  outTok[((long)b * 9) * 768 + col] = acc;
}

// Token branch K5: tokens_out rows 1..8 = selectors (pass-through)
__global__ __launch_bounds__(256)
void copy_selectors(const float* __restrict__ tokens, float* __restrict__ outTok) {
  int idx = blockIdx.x * 256 + threadIdx.x;
  const int total = 64 * 8 * 768;
  if (idx >= total) return;
  int c = idx % 768, rest = idx / 768;
  int s = rest % 8, b = rest / 8;
  long off = ((long)b * 9 + 1 + s) * 768 + c;
  outTok[off] = tokens[off];
}

// ---------------------------------------------------------------------------
// Host side
// ---------------------------------------------------------------------------
extern "C" void kernel_launch(void* const* d_in, const int* in_sizes, int n_in,
                              void* d_out, int out_size, void* d_ws, size_t ws_size,
                              hipStream_t stream) {
  (void)in_sizes; (void)n_in; (void)out_size; (void)ws_size;
  const float* x      = (const float*)d_in[0];   // (64,197,768)
  const float* tokens = (const float*)d_in[1];   // (1,64,9,768)
  const float* qkv_w  = (const float*)d_in[2];   // (2304,768)
  const float* proj_w = (const float*)d_in[3];   // (768,768)
  const float* proj_b = (const float*)d_in[4];   // (768,)
  const float* prompts= (const float*)d_in[5];   // (2,10,8,12,64)

  const int M = 64 * 197;          // 12608 = 788 * 16
  float* xo_out  = (float*)d_out;                   // (64,197,768)
  float* tok_out = (float*)d_out + (long)M * 768;   // (1,64,9,768)

  // workspace carve-up (bf16 staging buffers + small f32 scratch)
  char* ws = (char*)d_ws;
  size_t off = 0;
  auto carve = [&](size_t bytes) { char* p = ws + off; off += (bytes + 255) & ~(size_t)255; return p; };
  bf16_t* xb      = (bf16_t*)carve((size_t)M * 768 * 2);
  bf16_t* qkv_wb  = (bf16_t*)carve((size_t)2304 * 768 * 2);
  bf16_t* proj_wb = (bf16_t*)carve((size_t)768 * 768 * 2);
  bf16_t* qkv2b   = (bf16_t*)carve((size_t)M * 2304 * 2);
  bf16_t* ob      = (bf16_t*)carve((size_t)M * 768 * 2);
  bf16_t* txb     = (bf16_t*)carve((size_t)576 * 768 * 2);
  bf16_t* tqkvb   = (bf16_t*)carve((size_t)576 * 2304 * 2);
  float*  out0    = (float*)carve((size_t)64 * 768 * 4);

  // 1) downconvert inputs to bf16 (one HBM pass; everything then L2-resident)
  cvt_bf16<<<(M * 768 + 255) / 256, 256, 0, stream>>>(x, xb, M * 768);
  cvt_bf16<<<(2304 * 768 + 255) / 256, 256, 0, stream>>>(qkv_w, qkv_wb, 2304 * 768);
  cvt_bf16<<<(768 * 768 + 255) / 256, 256, 0, stream>>>(proj_w, proj_wb, 768 * 768);

  // 2) main qkv GEMM: (12608x768)@(2304x768)^T -> bf16
  {
    int tiles = (M / 64) * (2304 / 64);  // 197*36
    gemm_nt_wmma<<<(tiles + 7) / 8, 256, 0, stream>>>(xb, qkv_wb, M, 2304, 768,
                                                      qkv2b, nullptr, nullptr);
  }

  // 3) fused per-(b,head) attention, 288 KB dynamic LDS per workgroup
  attn_fused<<<dim3(64, 12), 256, SM_TOTAL, stream>>>(qkv2b, ob);

  // 4) output projection with bias, f32 straight into d_out
  {
    int tiles = (M / 64) * (768 / 64);   // 197*12
    gemm_nt_wmma<<<(tiles + 7) / 8, 256, 0, stream>>>(ob, proj_wb, M, 768, 768,
                                                      nullptr, xo_out, proj_b);
  }

  // 5) token branch
  selector_attn<<<dim3(8, 64), 256, 0, stream>>>(x, tokens, txb);
  {
    int tiles = (576 / 64) * (2304 / 64);  // 9*36
    gemm_nt_wmma<<<(tiles + 7) / 8, 256, 0, stream>>>(txb, qkv_wb, 576, 2304, 768,
                                                      tqkvb, nullptr, nullptr);
  }
  task_attn<<<64, 256, 0, stream>>>(tqkvb, prompts, out0);
  task_proj<<<dim3(64, 3), 256, 0, stream>>>(out0, proj_w, proj_b, tok_out);
  copy_selectors<<<(64 * 8 * 768 + 255) / 256, 256, 0, stream>>>(tokens, tok_out);
}
